// TGModel_62895501082806
// MI455X (gfx1250) — matrix operand order repaired and checked by reading, not compiled, not run on hardware
//
#include <hip/hip_runtime.h>

// ---------------------------------------------------------------------------
// Types for CDNA5 WMMA
// ---------------------------------------------------------------------------
typedef _Float16 v16h __attribute__((ext_vector_type(16)));
typedef _Float16 v8h  __attribute__((ext_vector_type(8)));
typedef _Float16 v4h  __attribute__((ext_vector_type(4)));
typedef float    v8f  __attribute__((ext_vector_type(8)));
typedef int      i32x4 __attribute__((ext_vector_type(4)));

#define B_   64
#define T_   120
#define MTOK (B_ * T_)        // 7680
#define DMODEL 1536
#define HEADS 12
#define DK   128
#define DFF  2048
#define HDIM 512
#define KSTEP 64
#define LDT  72               // padded LDS row stride (halves) for 64-k tiles

// Async global->LDS copy path (CDNA5), guarded so either toolchain compiles.
#if defined(__has_builtin)
# if __has_builtin(__builtin_amdgcn_global_load_async_to_lds_b128) && \
     __has_builtin(__builtin_amdgcn_s_wait_asynccnt)
#  define USE_ASYNC_B 1
# endif
#endif
#ifndef USE_ASYNC_B
# define USE_ASYNC_B 0
#endif

#if USE_ASYNC_B
typedef __attribute__((address_space(1))) i32x4* gptr_b128;   // __device__ int4*
typedef __attribute__((address_space(3))) i32x4* lptr_b128;   // __shared__ int4*

__device__ __forceinline__ void async_copy_b128(const _Float16* g, _Float16* l)
{
    // inttoptr casts: AS1 keeps the 64-bit global address, AS3 takes the
    // low 32 bits (flat->LDS truncation per ISA 10.2).
    __builtin_amdgcn_global_load_async_to_lds_b128(
        (gptr_b128)(unsigned long long)g,
        (lptr_b128)(unsigned)(unsigned long long)l,
        0, 0);
}
#endif

// ---------------------------------------------------------------------------
// fp32 [K,N] weight -> f16 transposed [N,K]
// ---------------------------------------------------------------------------
__global__ __launch_bounds__(256)
void k_cvt(const float* __restrict__ W, _Float16* __restrict__ Wt, int K, int N)
{
    const int idx = blockIdx.x * 256 + threadIdx.x;
    if (idx >= K * N) return;
    const int n = idx / K;
    const int k = idx - n * K;
    Wt[idx] = (_Float16)W[(size_t)k * N + n];
}

// ---------------------------------------------------------------------------
// WMMA GEMM:  C[M,N] = epi( A[M,K](f32) @ W[K,N] + bias ),  W given as Wt[N,K] f16
// Block: 256 thr = 8 waves.  C tile 128(M) x 128(N), K-step 64, double-buffered
// LDS; A register-staged (fp32->f16), B async-staged to LDS.
// Wave w owns rows [w*16, w*16+16) x all 128 cols -> 16 WMMAs / K-step.
// ---------------------------------------------------------------------------
__global__ __launch_bounds__(256)
void k_gemm(const float* __restrict__ A, const _Float16* __restrict__ Wt,
            const float* __restrict__ bias, const float* __restrict__ resid,
            float* __restrict__ C, int M, int N, int K, int relu)
{
    __shared__ _Float16 As[2][128 * LDT];
    __shared__ _Float16 Bs[2][128 * LDT];

    const int tid  = threadIdx.x;
    const int wave = tid >> 5;
    const int lane = tid & 31;
    const int lrow = lane & 15;
    const int lhi  = lane >> 4;
    const int m0   = blockIdx.x * 128;
    const int n0   = blockIdx.y * 128;

    // staging coordinates (constant per thread)
    int ar[8], ac[8];           // A: 128x64 fp32, 8 float4 units per thread
#pragma unroll
    for (int it = 0; it < 8; ++it) {
        const int idx = tid + it * 256;
        ar[it] = idx >> 4;             // 16 float4 per 64-wide row
        ac[it] = (idx & 15) << 2;
    }
    int br[4], bc[4];           // B: 128x64 f16, 4 v8h units per thread
#pragma unroll
    for (int it = 0; it < 4; ++it) {
        const int idx = tid + it * 256;
        br[it] = idx >> 3;             // 8 v8h per 64-wide row
        bc[it] = (idx & 7) << 3;
    }

    const v8f zero = {0.f, 0.f, 0.f, 0.f, 0.f, 0.f, 0.f, 0.f};
    v8f acc[8];
#pragma unroll
    for (int t = 0; t < 8; ++t) acc[t] = zero;

    float4 ra[8];
#if !USE_ASYNC_B
    v8h rb[4];
#endif

    // ---- prologue: stage tile 0
#pragma unroll
    for (int it = 0; it < 8; ++it)
        ra[it] = *(const float4*)(A + (size_t)(m0 + ar[it]) * K + ac[it]);
#if USE_ASYNC_B
#pragma unroll
    for (int it = 0; it < 4; ++it)
        async_copy_b128(Wt + (size_t)(n0 + br[it]) * K + bc[it],
                        &Bs[0][br[it] * LDT + bc[it]]);
#else
#pragma unroll
    for (int it = 0; it < 4; ++it)
        rb[it] = *(const v8h*)(Wt + (size_t)(n0 + br[it]) * K + bc[it]);
#endif
#pragma unroll
    for (int it = 0; it < 8; ++it) {
        v4h hv;
        hv[0] = (_Float16)ra[it].x; hv[1] = (_Float16)ra[it].y;
        hv[2] = (_Float16)ra[it].z; hv[3] = (_Float16)ra[it].w;
        *(v4h*)(&As[0][ar[it] * LDT + ac[it]]) = hv;
    }
#if USE_ASYNC_B
    __builtin_amdgcn_s_wait_asynccnt(0);
#else
#pragma unroll
    for (int it = 0; it < 4; ++it)
        *(v8h*)(&Bs[0][br[it] * LDT + bc[it]]) = rb[it];
#endif
    __syncthreads();

    const int nsteps = K >> 6;           // K-step 64
    const int mrow = wave * 16 + lrow;
    const int kh   = lhi * 8;

    for (int s = 0; s < nsteps; ++s) {
        const int cur = s & 1;
        const int nxt = cur ^ 1;
        const bool has_next = (s + 1) < nsteps;
        const int k0n = (s + 1) << 6;

        // ---- issue next tile's global traffic before this tile's math
        if (has_next) {
#pragma unroll
            for (int it = 0; it < 8; ++it)
                ra[it] = *(const float4*)(A + (size_t)(m0 + ar[it]) * K + k0n + ac[it]);
#if USE_ASYNC_B
#pragma unroll
            for (int it = 0; it < 4; ++it)
                async_copy_b128(Wt + (size_t)(n0 + br[it]) * K + k0n + bc[it],
                                &Bs[nxt][br[it] * LDT + bc[it]]);
#else
#pragma unroll
            for (int it = 0; it < 4; ++it)
                rb[it] = *(const v8h*)(Wt + (size_t)(n0 + br[it]) * K + k0n + bc[it]);
#endif
        }
        if (s + 2 < nsteps) {   // deep prefetch into L2 (global_prefetch_b8)
            const int k2 = (s + 2) << 6;
            __builtin_prefetch(A + (size_t)(m0 + ar[0]) * K + k2 + ac[0], 0, 1);
            __builtin_prefetch(Wt + (size_t)(n0 + br[0]) * K + k2 + bc[0], 0, 1);
        }

        // ---- math on current tile: 2 A fragments, 16 B fragments, 16 WMMAs
        const _Float16* arow = &As[cur][mrow * LDT];
        const v8h a0lo = *(const v8h*)(arow + kh);
        const v8h a0hi = *(const v8h*)(arow + 16 + kh);
        const v8h a1lo = *(const v8h*)(arow + 32 + kh);
        const v8h a1hi = *(const v8h*)(arow + 48 + kh);
        const v16h av0 = __builtin_shufflevector(a0lo, a0hi,
                          0,1,2,3,4,5,6,7,8,9,10,11,12,13,14,15);
        const v16h av1 = __builtin_shufflevector(a1lo, a1hi,
                          0,1,2,3,4,5,6,7,8,9,10,11,12,13,14,15);
#pragma unroll
        for (int t = 0; t < 8; ++t) {
            const _Float16* brow = &Bs[cur][(t * 16 + lrow) * LDT];
            const v8h b0lo = *(const v8h*)(brow + lhi * 8);
            const v8h b0hi = *(const v8h*)(brow + 16 + lhi * 8);
            const v16h bv0 = __builtin_shufflevector(b0lo, b0hi,
                              0,1,2,3,4,5,6,7,8,9,10,11,12,13,14,15);
            acc[t] = __builtin_amdgcn_wmma_f32_16x16x32_f16(
                         false, av0, false, bv0, (short)0, acc[t], false, false);
            const v8h b1lo = *(const v8h*)(brow + 32 + lhi * 8);
            const v8h b1hi = *(const v8h*)(brow + 48 + lhi * 8);
            const v16h bv1 = __builtin_shufflevector(b1lo, b1hi,
                              0,1,2,3,4,5,6,7,8,9,10,11,12,13,14,15);
            acc[t] = __builtin_amdgcn_wmma_f32_16x16x32_f16(
                         false, av1, false, bv1, (short)0, acc[t], false, false);
        }

        // ---- commit next tile into the other LDS buffer
        if (has_next) {
#pragma unroll
            for (int it = 0; it < 8; ++it) {
                v4h hv;
                hv[0] = (_Float16)ra[it].x; hv[1] = (_Float16)ra[it].y;
                hv[2] = (_Float16)ra[it].z; hv[3] = (_Float16)ra[it].w;
                *(v4h*)(&As[nxt][ar[it] * LDT + ac[it]]) = hv;
            }
#if USE_ASYNC_B
            __builtin_amdgcn_s_wait_asynccnt(0);
#else
#pragma unroll
            for (int it = 0; it < 4; ++it)
                *(v8h*)(&Bs[nxt][br[it] * LDT + bc[it]]) = rb[it];
#endif
        }
        __syncthreads();
    }

    // ---- epilogue: +bias, optional +resid, optional relu
#pragma unroll
    for (int t = 0; t < 8; ++t) {
        const int n  = n0 + t * 16 + lrow;
        const float bv = bias ? bias[n] : 0.f;
#pragma unroll
        for (int v = 0; v < 8; ++v) {
            const int m = m0 + wave * 16 + (lhi << 3) + v;   // m = v + 8*(L/16)
            float x = acc[t][v] + bv;
            if (resid) x += resid[(size_t)m * N + n];
            if (relu)  x = fmaxf(x, 0.f);
            C[(size_t)m * N + n] = x;
        }
    }
}

// ---------------------------------------------------------------------------
// PE + concat: h[bt, {0,512,1024}+c] = {s,o,t}[bt,c] + pe[tta][c]
// ---------------------------------------------------------------------------
__global__ __launch_bounds__(512)
void k_pe_concat(const float* __restrict__ s, const float* __restrict__ o,
                 const float* __restrict__ tg, const int* __restrict__ input_len,
                 float* __restrict__ h)
{
    const int bt = blockIdx.x;
    const int c  = threadIdx.x;
    const int b  = bt / T_;
    const int t  = bt - b * T_;
    int tta = input_len[b] - t;
    if (tta < 0)   tta = 0;
    if (tta > 120) tta = 120;
    float pe = 0.f;
    if (tta > 0) {
        const int ce = c & ~1;
        const float ang = (float)tta * __expf(-(float)ce * (9.210340371976184f / 512.f));
        pe = (c & 1) ? __cosf(ang) : __sinf(ang);
    }
    const size_t base = (size_t)bt * DMODEL;
    const size_t sb   = (size_t)bt * HDIM + c;
    h[base + c]          = s[sb]  + pe;
    h[base + HDIM + c]   = o[sb]  + pe;
    h[base + 2*HDIM + c] = tg[sb] + pe;
}

// ---------------------------------------------------------------------------
// LayerNorm (torch-style: unbiased var, denom = std + eps), D = 1536
// ---------------------------------------------------------------------------
__global__ __launch_bounds__(256)
void k_ln(const float* __restrict__ x, const float* __restrict__ g,
          const float* __restrict__ be, float* __restrict__ y)
{
    __shared__ float rs[256], rq[256];
    const int row = blockIdx.x;
    const int tid = threadIdx.x;
    const float* xr = x + (size_t)row * DMODEL;
    float s = 0.f, sq = 0.f;
    for (int c = tid; c < DMODEL; c += 256) { const float v = xr[c]; s += v; sq += v * v; }
    rs[tid] = s; rq[tid] = sq; __syncthreads();
    for (int o = 128; o > 0; o >>= 1) {
        if (tid < o) { rs[tid] += rs[tid + o]; rq[tid] += rq[tid + o]; }
        __syncthreads();
    }
    const float mean = rs[0] * (1.f / DMODEL);
    const float var  = fmaxf((rq[0] - (float)DMODEL * mean * mean) / (DMODEL - 1), 0.f);
    const float inv  = 1.f / (sqrtf(var) + 1e-6f);
    float* yr = y + (size_t)row * DMODEL;
    for (int c = tid; c < DMODEL; c += 256)
        yr[c] = g[c] * (xr[c] - mean) * inv + be[c];
}

// ---------------------------------------------------------------------------
// Attention: one 128-thread block per (b, head, i) row.
// scores = (q.k / sqrt(128)) * GaussBand, softmax over ALL 120 cols, then @V.
// ---------------------------------------------------------------------------
__global__ __launch_bounds__(128)
void k_attn(const float* __restrict__ Q, const float* __restrict__ Kx,
            const float* __restrict__ V, const int* __restrict__ len,
            float* __restrict__ O)
{
    __shared__ float qrow[DK];
    __shared__ float attn[128];
    __shared__ float red[128];
    const int tid = threadIdx.x;
    const int i  = blockIdx.x % T_;
    const int hh = (blockIdx.x / T_) % HEADS;
    const int b  = blockIdx.x / (T_ * HEADS);
    const int L  = len[b];
    const size_t rowQ = (size_t)(b * T_ + i) * DMODEL + hh * DK;
    qrow[tid] = Q[rowQ + tid];
    __syncthreads();

    const int j = tid;
    float sc = -3.0e38f;
    if (j < T_) {
        const float* kr = Kx + (size_t)(b * T_ + j) * DMODEL + hh * DK;
        float d = 0.f;
#pragma unroll 4
        for (int c = 0; c < DK; ++c) d += qrow[c] * kr[c];
        float m = 0.f;
        const int dd = i - j;
        if (dd * dd <= 225 && i < L && j < L)
            m = __expf(-0.5f * (float)(dd * dd)) * 0.3989422804014327f;
        sc = d * 0.08838834764831845f * m;   // 1/sqrt(128)
    }
    red[tid] = sc; __syncthreads();
    for (int o = 64; o > 0; o >>= 1) {
        if (tid < o) red[tid] = fmaxf(red[tid], red[tid + o]);
        __syncthreads();
    }
    const float mx = red[0];
    __syncthreads();
    const float e = (j < T_) ? __expf(sc - mx) : 0.f;
    red[tid] = e; __syncthreads();
    for (int o = 64; o > 0; o >>= 1) {
        if (tid < o) red[tid] += red[tid + o];
        __syncthreads();
    }
    attn[tid] = e / red[0];
    __syncthreads();

    float acc = 0.f;
    const size_t vbase = (size_t)(b * T_) * DMODEL + hh * DK + tid;
    for (int jj = 0; jj < T_; ++jj)
        acc += attn[jj] * V[vbase + (size_t)jj * DMODEL];
    O[rowQ + tid] = acc;
}

// ---------------------------------------------------------------------------
// Host orchestration
// ---------------------------------------------------------------------------
extern "C" void kernel_launch(void* const* d_in, const int* in_sizes, int n_in,
                              void* d_out, int out_size, void* d_ws, size_t ws_size,
                              hipStream_t stream)
{
    (void)in_sizes; (void)n_in; (void)out_size; (void)ws_size;

    const float* x_state  = (const float*)d_in[0];
    const float* x_offset = (const float*)d_in[1];
    const float* x_target = (const float*)d_in[2];
    const int*   input_len = (const int*)d_in[4];            // d_in[3] = mask (unused)
    const int ENC_BASE[3] = {5, 11, 17};                     // state, offset, target
    const int LAYER_BASE  = 23;                              // +16 per layer
    const int NORM_A = 87, NORM_B = 88, GEN_W = 89, GEN_B = 90;

    // ---- workspace carve-out
    char* ws = (char*)d_ws;
    size_t off = 0;
    auto f32buf = [&](size_t n) { float* p = (float*)(ws + off); off += n * 4; return p; };
    auto f16buf = [&](size_t n) { _Float16* p = (_Float16*)(ws + off); off += n * 2; return p; };

    float* h   = f32buf((size_t)MTOK * DMODEL);
    float* xln = f32buf((size_t)MTOK * DMODEL);
    float* q   = f32buf((size_t)MTOK * DMODEL);   // also enc state out
    float* kk  = f32buf((size_t)MTOK * DMODEL);   // also enc offset out
    float* vv  = f32buf((size_t)MTOK * DMODEL);   // also enc target out
    float* ao  = f32buf((size_t)MTOK * DMODEL);
    float* ffo = f32buf((size_t)MTOK * DFF);
    float* t1  = f32buf((size_t)MTOK * HDIM);
    float* t2  = f32buf((size_t)MTOK * HDIM);

    // ---- convert all weights fp32[K,N] -> f16 transposed [N,K]
    auto cvt = [&](int widx, int K, int N) -> _Float16* {
        _Float16* dst = f16buf((size_t)K * N);
        const int total = K * N;
        k_cvt<<<(total + 255) / 256, 256, 0, stream>>>((const float*)d_in[widx], dst, K, N);
        return dst;
    };

    _Float16* encW[3][3];
    for (int s = 0; s < 3; ++s) {
        encW[s][0] = cvt(ENC_BASE[s] + 0, 256, HDIM);
        encW[s][1] = cvt(ENC_BASE[s] + 2, HDIM, HDIM);
        encW[s][2] = cvt(ENC_BASE[s] + 4, HDIM, HDIM);
    }
    _Float16 *wqT[4], *wkT[4], *wvT[4], *woT[4], *ff1T[4], *ff2T[4];
    for (int l = 0; l < 4; ++l) {
        const int base = LAYER_BASE + l * 16;
        wqT[l]  = cvt(base + 0,  DMODEL, DMODEL);
        wkT[l]  = cvt(base + 2,  DMODEL, DMODEL);
        wvT[l]  = cvt(base + 4,  DMODEL, DMODEL);
        woT[l]  = cvt(base + 6,  DMODEL, DMODEL);
        ff1T[l] = cvt(base + 8,  DMODEL, DFF);
        ff2T[l] = cvt(base + 10, DFF,    DMODEL);
    }
    _Float16* genT = cvt(GEN_W, DMODEL, 256);

    auto gemm = [&](const float* A, const _Float16* Wt, int bidx,
                    const float* resid, float* C, int N, int K, int relu) {
        dim3 g(MTOK / 128, N / 128);
        k_gemm<<<g, 256, 0, stream>>>(A, Wt, (const float*)d_in[bidx], resid, C,
                                      MTOK, N, K, relu);
    };

    // ---- three FC encoders (relu after every layer)
    const float* enc_in[3] = {x_state, x_offset, x_target};
    float* enc_out[3] = {q, kk, vv};
    for (int s = 0; s < 3; ++s) {
        gemm(enc_in[s], encW[s][0], ENC_BASE[s] + 1, nullptr, t1, HDIM, 256,  1);
        gemm(t1,        encW[s][1], ENC_BASE[s] + 3, nullptr, t2, HDIM, HDIM, 1);
        gemm(t2,        encW[s][2], ENC_BASE[s] + 5, nullptr, enc_out[s], HDIM, HDIM, 1);
    }

    // ---- positional encoding + concat -> h (B,T,1536)
    k_pe_concat<<<MTOK, 512, 0, stream>>>(q, kk, vv, input_len, h);

    // ---- transformer layers
    for (int l = 0; l < 4; ++l) {
        const int base = LAYER_BASE + l * 16;
        k_ln<<<MTOK, 256, 0, stream>>>(h, (const float*)d_in[base + 12],
                                       (const float*)d_in[base + 13], xln);
        gemm(xln, wqT[l], base + 1, nullptr, q,  DMODEL, DMODEL, 0);
        gemm(xln, wkT[l], base + 3, nullptr, kk, DMODEL, DMODEL, 0);
        gemm(xln, wvT[l], base + 5, nullptr, vv, DMODEL, DMODEL, 0);
        k_attn<<<B_ * HEADS * T_, 128, 0, stream>>>(q, kk, vv, input_len, ao);
        gemm(ao, woT[l], base + 7, h, h, DMODEL, DMODEL, 0);     // h += mha(...)
        k_ln<<<MTOK, 256, 0, stream>>>(h, (const float*)d_in[base + 14],
                                       (const float*)d_in[base + 15], xln);
        gemm(xln, ff1T[l], base + 9,  nullptr, ffo, DFF,    DMODEL, 1);
        gemm(ffo, ff2T[l], base + 11, h,       h,   DMODEL, DFF,    0);  // h += ff(...)
    }

    // ---- final norm + generator
    k_ln<<<MTOK, 256, 0, stream>>>(h, (const float*)d_in[NORM_A],
                                   (const float*)d_in[NORM_B], xln);
    gemm(xln, genT, GEN_B, nullptr, (float*)d_out, 256, DMODEL, 0);
}
